// AsymAttention_10170482557322
// MI455X (gfx1250) — matrix operand order repaired and checked by reading, not compiled, not run on hardware
//
#include <hip/hip_runtime.h>

// ---------------------------------------------------------------------------
// Problem constants
// ---------------------------------------------------------------------------
#define BB 64
#define NN 196
#define LL 49
#define MM 8
#define DD 768
#define HH 8
#define DH 96
#define NPAD 208      // 13 * 16
#define KPAD 224      // 7 * 32 (K padding for P@V)
static constexpr float SCALE = 0.1020620726159658f;   // 96^-0.5

typedef __attribute__((ext_vector_type(16))) _Float16 v16h;
typedef __attribute__((ext_vector_type(8)))  float    v8f;

// ---------------------------------------------------------------------------
// WMMA fragment loaders (wave32, v_wmma_f32_16x16x32_f16 layouts per ISA 7.12.2)
// A 16x32 f16: lane = M (0..15, both halves); lanes 0-15 hold K 0-7 & 16-23,
//              lanes 16-31 hold K 8-15 & 24-31, packed 2 per VGPR.
// B 32x16 f16 read from an N-major LDS buffer BT[n][k] (element B[k][n]):
//              lane = N; lanes 0-15 hold K 0-7 & 16-23, lanes 16-31 hold K 8-15 & 24-31.
// ---------------------------------------------------------------------------
__device__ __forceinline__ v16h load_frag_a(const _Float16* lds, int rowBase,
                                            int stride, int kBase, int lane) {
  const int m = lane & 15, half = lane >> 4;
  const _Float16* p = lds + (rowBase + m) * stride + kBase;
  union { v16h v; unsigned u[8]; } a;
#pragma unroll
  for (int j = 0; j < 8; ++j) {
    const int kk = (j < 4) ? (2 * j + half * 8) : (16 + 2 * (j - 4) + half * 8);
    a.u[j] = *reinterpret_cast<const unsigned*>(p + kk);
  }
  return a.v;
}

__device__ __forceinline__ v16h load_frag_bt(const _Float16* ldsBT, int nBase,
                                             int stride, int kBase, int lane) {
  const int n = lane & 15, half = lane >> 4;
  const _Float16* p = ldsBT + (nBase + n) * stride + kBase + half * 8;
  union { v16h v; unsigned u[8]; } b;
#pragma unroll
  for (int j = 0; j < 8; ++j)
    b.u[j] = *reinterpret_cast<const unsigned*>(p + ((j < 4) ? (2 * j) : (16 + 2 * (j - 4))));
  return b.v;
}

// ---------------------------------------------------------------------------
// Kernel 1: pos[b][n] = rank among kept (mask==0) positions, else -1.
// stable argsort of 0/1 mask keeps zeros in original order => rank map.
// ---------------------------------------------------------------------------
__global__ __launch_bounds__(64) void build_pos_kernel(const int* __restrict__ mask,
                                                       int* __restrict__ pos) {
  const int b = threadIdx.x;
  if (b >= BB) return;
  int cnt = 0;
  for (int n = 0; n < NN; ++n) {
    const int kept = (mask[b * NN + n] == 0);
    pos[b * NN + n] = kept ? cnt : -1;
    cnt += kept;
  }
}

// ---------------------------------------------------------------------------
// Kernel 2: C[rows,768] = A[rows,768] @ W[768,768] + bias   (f16 WMMA, f32 acc)
// Block tile 128x128, 256 threads = 8 waves, each wave 32x64 (2x4 WMMA tiles).
// Software-pipelined double buffering: global loads for step k+1 are issued
// into VGPRs before the 8-WMMA burst of step k (latency hidden by the matrix
// pipe), then converted+stored into the alternate LDS buffer. One barrier per
// K-step. K+2 tiles are prefetched (global_prefetch_b8) into cache.
// rows % 128 == 0, so no bounds checks.
// ---------------------------------------------------------------------------
__global__ __launch_bounds__(256) void gemm_bias_kernel(
    const float* __restrict__ A, const float* __restrict__ W,
    const float* __restrict__ bias, float* __restrict__ C) {
  __shared__ _Float16 sA[2][128 * 32];   // A tile, row-major, stride 32
  __shared__ _Float16 sWt[2][128 * 32];  // W tile transposed: [col][k], stride 32

  const int tid  = threadIdx.x;
  const int lane = tid & 31;
  const int wave = tid >> 5;
  const int row0 = blockIdx.y * 128;
  const int col0 = blockIdx.x * 128;
  const int wm = wave & 3;   // row group  (4 x 32 rows)
  const int wn = wave >> 2;  // col group  (2 x 64 cols)

  // per-thread staging coordinates (constant across K-steps)
  const int ar = tid >> 3, ac4 = tid & 7;     // A: idx = tid + i*256 -> r=idx>>3
  const int wk = tid >> 5, wc4 = tid & 31;    // W: kk=idx>>5, c4=idx&31

  v8f acc[2][4];
#pragma unroll
  for (int i = 0; i < 2; ++i)
#pragma unroll
    for (int j = 0; j < 4; ++j) acc[i][j] = {};

  float4 ra[4], rw[4];

  auto load_tiles = [&](int k0) {
#pragma unroll
    for (int i = 0; i < 4; ++i) {
      const int idx = tid + i * 256;
      const int r = idx >> 3, c4 = idx & 7;
      ra[i] = *reinterpret_cast<const float4*>(
          A + (size_t)(row0 + r) * DD + k0 + c4 * 4);
    }
#pragma unroll
    for (int i = 0; i < 4; ++i) {
      const int idx = tid + i * 256;
      const int kk = idx >> 5, c4 = idx & 31;
      rw[i] = *reinterpret_cast<const float4*>(
          W + (size_t)(k0 + kk) * DD + col0 + c4 * 4);
    }
  };

  auto store_tiles = [&](int buf) {
#pragma unroll
    for (int i = 0; i < 4; ++i) {
      const int idx = tid + i * 256;
      const int r = idx >> 3, c4 = idx & 7;
      union { _Float16 h[4]; unsigned long long u; } p;
      p.h[0] = (_Float16)ra[i].x; p.h[1] = (_Float16)ra[i].y;
      p.h[2] = (_Float16)ra[i].z; p.h[3] = (_Float16)ra[i].w;
      *reinterpret_cast<unsigned long long*>(&sA[buf][r * 32 + c4 * 4]) = p.u;
    }
#pragma unroll
    for (int i = 0; i < 4; ++i) {
      const int idx = tid + i * 256;
      const int kk = idx >> 5, c4 = idx & 31;
      sWt[buf][(c4 * 4 + 0) * 32 + kk] = (_Float16)rw[i].x;
      sWt[buf][(c4 * 4 + 1) * 32 + kk] = (_Float16)rw[i].y;
      sWt[buf][(c4 * 4 + 2) * 32 + kk] = (_Float16)rw[i].z;
      sWt[buf][(c4 * 4 + 3) * 32 + kk] = (_Float16)rw[i].w;
    }
  };

  // prologue: stage K-step 0 into buffer 0
  load_tiles(0);
  store_tiles(0);

  constexpr int KSTEPS = DD / 32;   // 24
  for (int k = 0; k < KSTEPS; ++k) {
    __syncthreads();
    const int buf = k & 1;

    // issue next step's global loads before the WMMA burst
    if (k + 1 < KSTEPS) load_tiles((k + 1) * 32);
    // warm L2/WGP$ two steps ahead (global_prefetch_b8)
    if (k + 2 < KSTEPS) {
      __builtin_prefetch(A + (size_t)(row0 + ar) * DD + (k + 2) * 32 + ac4 * 4, 0, 0);
      __builtin_prefetch(W + (size_t)((k + 2) * 32 + wk) * DD + col0 + wc4 * 4, 0, 0);
    }

    v16h afrag[2], bfrag[4];
#pragma unroll
    for (int mt = 0; mt < 2; ++mt)
      afrag[mt] = load_frag_a(sA[buf], wm * 32 + mt * 16, 32, 0, lane);
#pragma unroll
    for (int nt = 0; nt < 4; ++nt)
      bfrag[nt] = load_frag_bt(sWt[buf], wn * 64 + nt * 16, 32, 0, lane);
#pragma unroll
    for (int mt = 0; mt < 2; ++mt)
#pragma unroll
      for (int nt = 0; nt < 4; ++nt)
        acc[mt][nt] = __builtin_amdgcn_wmma_f32_16x16x32_f16(
            false, afrag[mt], false, bfrag[nt], (short)0, acc[mt][nt],
            false, false);

    // convert + store next tile into the other buffer (after compute so the
    // global-load wait overlaps the WMMA burst)
    if (k + 1 < KSTEPS) store_tiles(buf ^ 1);
  }

  const int half = lane >> 4, ln = lane & 15;
#pragma unroll
  for (int mt = 0; mt < 2; ++mt)
#pragma unroll
    for (int nt = 0; nt < 4; ++nt) {
      const int col = col0 + wn * 64 + nt * 16 + ln;
      const float bval = bias[col];
#pragma unroll
      for (int r = 0; r < 8; ++r) {
        const int row = row0 + wm * 32 + mt * 16 + r + half * 8;
        C[(size_t)row * DD + col] = acc[mt][nt][r] + bval;
      }
    }
}

// ---------------------------------------------------------------------------
// Kernel 3: attention per (b,h). One workgroup (8 waves) per head.
// LDS layout (319,104 B total, fits CDNA5's 320 KB/WGP):
//   [0)        sQ  208x96 f16 (39936)  \ later aliased by
//   [39936)    sK  208x96 f16 (39936)  /  sP 208x224 f16 (93184)
//   [94208)    sS  208x196 f32 (163072)
//   [257280)   sVt 96x224 f16 (43008)   (V transposed: [d][n])
//   [300288)   sSimO 49x96 f32 (18816)
// ---------------------------------------------------------------------------
#define OFF_K    39936
#define OFF_S    94208
#define OFF_V    257280
#define OFF_SIMO 300288
#define ATTN_LDS 319104

__global__ __launch_bounds__(256) void attn_kernel(
    const float* __restrict__ Qg, const float* __restrict__ Kg,
    const float* __restrict__ Vg, const float* __restrict__ Ksim,
    const float* __restrict__ Vsim, const int* __restrict__ pos,
    float* __restrict__ Og) {
  extern __shared__ char smem[];
  _Float16* sQ   = reinterpret_cast<_Float16*>(smem);
  _Float16* sK   = reinterpret_cast<_Float16*>(smem + OFF_K);
  _Float16* sP   = reinterpret_cast<_Float16*>(smem);          // aliases Q+K
  float*    sS   = reinterpret_cast<float*>(smem + OFF_S);
  _Float16* sVt  = reinterpret_cast<_Float16*>(smem + OFF_V);
  float*    sSim = reinterpret_cast<float*>(smem + OFF_SIMO);

  const int h = blockIdx.x;
  const int b = blockIdx.y;
  const int tid  = threadIdx.x;
  const int lane = tid & 31;
  const int wave = tid >> 5;
  const int half = lane >> 4, ln = lane & 15;
  const size_t baseRow = (size_t)b * NN;

  // ---- P0: stage Q, K (row-major) and V (transposed), zero sim accumulator
  for (int idx = tid; idx < NPAD * DH; idx += 256) {
    const int n = idx / DH, d = idx % DH;
    float q = 0.f, k = 0.f;
    if (n < NN) {
      const size_t g = (baseRow + n) * DD + h * DH + d;
      q = Qg[g]; k = Kg[g];
    }
    sQ[idx] = (_Float16)q;
    sK[idx] = (_Float16)k;
  }
  for (int idx = tid; idx < DH * KPAD; idx += 256) {
    const int d = idx / KPAD, n = idx % KPAD;
    float v = (n < NN) ? Vg[(baseRow + n) * DD + h * DH + d] : 0.f;
    sVt[idx] = (_Float16)v;
  }
  for (int idx = tid; idx < LL * DH; idx += 256) sSim[idx] = 0.f;
  __syncthreads();

  // ---- P1: S = Q @ K^T * SCALE   (13x13 tiles of 16x16, K = 96 = 3x32)
  for (int t = wave; t < 13 * 13; t += 8) {
    const int ti = t / 13, tj = t % 13;
    v8f acc = {};
#pragma unroll
    for (int ks = 0; ks < 3; ++ks) {
      v16h a = load_frag_a(sQ, ti * 16, DH, ks * 32, lane);
      v16h bb = load_frag_bt(sK, tj * 16, DH, ks * 32, lane);
      acc = __builtin_amdgcn_wmma_f32_16x16x32_f16(false, a, false, bb,
                                                   (short)0, acc, false, false);
    }
    const int col = tj * 16 + ln;
    if (col < NN) {
#pragma unroll
      for (int r = 0; r < 8; ++r) {
        const int row = ti * 16 + r + half * 8;
        sS[row * NN + col] = acc[r] * SCALE;
      }
    }
  }
  __syncthreads();

  // ---- P2: softmax over [196 self | 8 sim] logits per row; sim output to LDS
  for (int row = wave; row < NN; row += 8) {
    const int l = pos[b * NN + row];
    float mx = -3.0e38f;
    for (int c = lane; c < NN; c += 32) mx = fmaxf(mx, sS[row * NN + c]);
    float sl = -3.0e38f;
    if (l >= 0 && lane < MM) {
      const float* kp = Ksim + (((size_t)b * LL + l) * MM + lane) * DD + h * DH;
      float s = 0.f;
      for (int d = 0; d < DH; ++d) s += (float)sQ[row * DH + d] * kp[d];
      sl = s * SCALE;
      mx = fmaxf(mx, sl);
    }
#pragma unroll
    for (int off = 16; off; off >>= 1) mx = fmaxf(mx, __shfl_xor(mx, off, 32));

    float sum = 0.f;
    for (int c = lane; c < NN; c += 32) sum += __expf(sS[row * NN + c] - mx);
    float sp = 0.f;
    if (l >= 0 && lane < MM) { sp = __expf(sl - mx); sum += sp; }
#pragma unroll
    for (int off = 16; off; off >>= 1) sum += __shfl_xor(sum, off, 32);
    const float inv = 1.f / sum;

    for (int c = lane; c < NN; c += 32)
      sS[row * NN + c] = __expf(sS[row * NN + c] - mx) * inv;

    if (l >= 0) {
      float pm[MM];
#pragma unroll
      for (int m = 0; m < MM; ++m) pm[m] = __shfl(sp, m, 32) * inv;
      const float* vp = Vsim + (((size_t)b * LL + l) * MM) * DD + h * DH;
      for (int d = lane; d < DH; d += 32) {
        float o = 0.f;
#pragma unroll
        for (int m = 0; m < MM; ++m) o += pm[m] * vp[(size_t)m * DD + d];
        sSim[l * DH + d] = o;   // unique l per kept row -> no conflicts
      }
    }
  }
  __syncthreads();

  // ---- P3: convert probabilities to f16 with K padded to 224 (zero fill)
  for (int idx = tid; idx < NPAD * KPAD; idx += 256) {
    const int row = idx / KPAD, col = idx % KPAD;
    const float v = (row < NN && col < NN) ? sS[row * NN + col] : 0.f;
    sP[idx] = (_Float16)v;
  }
  __syncthreads();

  // ---- P4: O = P @ V  (13x6 tiles, K = 224 = 7x32), add sim contribution
  for (int t = wave; t < 13 * 6; t += 8) {
    const int ti = t / 6, tj = t % 6;
    v8f acc = {};
#pragma unroll
    for (int ks = 0; ks < 7; ++ks) {
      v16h a = load_frag_a(sP, ti * 16, KPAD, ks * 32, lane);
      v16h bb = load_frag_bt(sVt, tj * 16, KPAD, ks * 32, lane);
      acc = __builtin_amdgcn_wmma_f32_16x16x32_f16(false, a, false, bb,
                                                   (short)0, acc, false, false);
    }
    const int col = tj * 16 + ln;
#pragma unroll
    for (int r = 0; r < 8; ++r) {
      const int row = ti * 16 + r + half * 8;
      if (row < NN) {
        const int l = pos[b * NN + row];
        const float add = (l >= 0) ? sSim[l * DH + col] : 0.f;
        Og[(baseRow + row) * DD + h * DH + col] = acc[r] + add;
      }
    }
  }
}

// ---------------------------------------------------------------------------
// Host launcher
// ---------------------------------------------------------------------------
extern "C" void kernel_launch(void* const* d_in, const int* in_sizes, int n_in,
                              void* d_out, int out_size, void* d_ws, size_t ws_size,
                              hipStream_t stream) {
  const float* x    = (const float*)d_in[0];
  const int*   mask = (const int*)  d_in[1];
  const float* sim  = (const float*)d_in[2];
  const float* Wq   = (const float*)d_in[3];
  const float* bq   = (const float*)d_in[4];
  const float* Wk   = (const float*)d_in[5];
  const float* bk   = (const float*)d_in[6];
  const float* Wv   = (const float*)d_in[7];
  const float* bv   = (const float*)d_in[8];
  const float* Wp   = (const float*)d_in[9];
  const float* bp   = (const float*)d_in[10];

  constexpr size_t QSZ   = (size_t)BB * NN * DD;            // 9,633,792
  constexpr size_t SIMSZ = (size_t)BB * LL * MM * DD;       // 19,267,584

  float* ws     = (float*)d_ws;
  float* wsQ    = ws;
  float* wsK    = wsQ + QSZ;
  float* wsV    = wsK + QSZ;
  float* wsO    = wsV + QSZ;
  float* wsKsim = wsO + QSZ;
  int*   wsPos  = (int*)(wsKsim + SIMSZ);

  float* outMain = (float*)d_out;          // (B,N,D)
  float* outVsim = outMain + QSZ;          // (B,L,M,D) == sim@Wv+bv

  build_pos_kernel<<<1, 64, 0, stream>>>(mask, wsPos);

  const dim3 gX(DD / 128, (BB * NN) / 128);          // 6 x 98
  const dim3 gS(DD / 128, (BB * LL * MM) / 128);     // 6 x 196

  gemm_bias_kernel<<<gX, 256, 0, stream>>>(x,   Wq, bq, wsQ);
  gemm_bias_kernel<<<gX, 256, 0, stream>>>(x,   Wk, bk, wsK);
  gemm_bias_kernel<<<gX, 256, 0, stream>>>(x,   Wv, bv, wsV);
  gemm_bias_kernel<<<gS, 256, 0, stream>>>(sim, Wk, bk, wsKsim);
  gemm_bias_kernel<<<gS, 256, 0, stream>>>(sim, Wv, bv, outVsim);

  attn_kernel<<<dim3(HH, BB), 256, ATTN_LDS, stream>>>(
      wsQ, wsK, wsV, wsKsim, outVsim, wsPos, wsO);

  gemm_bias_kernel<<<gX, 256, 0, stream>>>(wsO, Wp, bp, outMain);
}
